// TemporalTransformer_81192061764333
// MI455X (gfx1250) — matrix operand order repaired and checked by reading, not compiled, not run on hardware
//
#include <hip/hip_runtime.h>

// ---------------------------------------------------------------------------
// TemporalTransformer forward, CDNA5 (gfx1250) wave32 WMMA implementation.
// Matmuls: v_wmma_f32_16x16x32_bf16. GEMM A-tile staging uses the CDNA5
// async global->LDS path (ASYNCcnt) with double-buffered LDS.
// ---------------------------------------------------------------------------

#define DIMC   1024
#define TLEN   1024
#define BATCH  4
#define NHEAD  16
#define FFDIM  4096
#define DEPTHC 4
#define MROWS  (BATCH * TLEN)   // 4096 token rows

typedef __attribute__((ext_vector_type(16))) __bf16 v16bf;
typedef __attribute__((ext_vector_type(8)))  float  v8f;

union Frag { v16bf v; unsigned u[8]; uint4 q[2]; };

__device__ __forceinline__ unsigned short f2bf(float x) {
  union { float f; unsigned u; } c; c.f = x;
  unsigned r = c.u + 0x7FFFu + ((c.u >> 16) & 1u);   // round-to-nearest-even
  return (unsigned short)(r >> 16);
}

__device__ __forceinline__ v8f vzero8() {
  v8f z = {0.f, 0.f, 0.f, 0.f, 0.f, 0.f, 0.f, 0.f};
  return z;
}

__device__ __forceinline__ v8f wmma_bf(const Frag& a, const Frag& b, v8f c) {
  // D = A(16x32 bf16) * B(32x16 bf16) + C(16x16 f32)
  return __builtin_amdgcn_wmma_f32_16x16x32_bf16(
      false, a.v, false, b.v, (short)0, c, false, false);
}

// Async copy of 16 bytes global -> LDS (per lane), tracked by ASYNCcnt.
__device__ __forceinline__ void async_copy_b128(void* lds_dst, const void* gsrc) {
  unsigned lds = (unsigned)(uintptr_t)lds_dst;            // low 32b = LDS offset
  unsigned long long ga = (unsigned long long)(uintptr_t)gsrc;
  asm volatile("global_load_async_to_lds_b128 %0, %1, off"
               :: "v"(lds), "v"(ga) : "memory");
}
__device__ __forceinline__ void wait_async0() {
  asm volatile("s_wait_asynccnt 0" ::: "memory");
}

// ---------------------------------------------------------------------------
// fp32 -> bf16 conversion
// ---------------------------------------------------------------------------
__global__ __launch_bounds__(256) void convert_f32_bf16(
    const float* __restrict__ in, unsigned short* __restrict__ out, int n) {
  int i = blockIdx.x * 256 + threadIdx.x;
  int stride = gridDim.x * 256;
  for (; i < n; i += stride) out[i] = f2bf(in[i]);
}

// ---------------------------------------------------------------------------
// LayerNorm: one block (8 waves) per row. OUT_BF16=1 -> bf16 out buffer,
// OUT_BF16=0 -> fp32 out (may alias input: per-element read precedes write).
// ---------------------------------------------------------------------------
template <int OUT_BF16>
__global__ __launch_bounds__(256) void layernorm_kernel(
    const float* __restrict__ x, const float* __restrict__ g,
    const float* __restrict__ b, float* __restrict__ outf,
    unsigned short* __restrict__ outb, int dim) {
  __shared__ float red[2][8];
  __shared__ float sh_mean, sh_rstd;
  const int row  = blockIdx.x;
  const int tid  = threadIdx.x;
  const int lane = tid & 31, wid = tid >> 5;
  const float* xr = x + (size_t)row * dim;

  float s = 0.f, ss = 0.f;
  for (int i = tid; i < dim; i += 256) {
    float v = xr[i];
    s += v; ss += v * v;
  }
  #pragma unroll
  for (int off = 16; off; off >>= 1) {
    s  += __shfl_xor(s,  off, 32);
    ss += __shfl_xor(ss, off, 32);
  }
  if (lane == 0) { red[0][wid] = s; red[1][wid] = ss; }
  __syncthreads();
  if (wid == 0) {
    float a = (lane < 8) ? red[0][lane] : 0.f;
    float c = (lane < 8) ? red[1][lane] : 0.f;
    #pragma unroll
    for (int off = 4; off; off >>= 1) {
      a += __shfl_xor(a, off, 32);
      c += __shfl_xor(c, off, 32);
    }
    if (lane == 0) {
      float mean = a / (float)dim;
      float var  = c / (float)dim - mean * mean;
      sh_mean = mean;
      sh_rstd = rsqrtf(var + 1e-5f);
    }
  }
  __syncthreads();
  const float mean = sh_mean, rstd = sh_rstd;
  for (int i = tid; i < dim; i += 256) {
    float v = (xr[i] - mean) * rstd * g[i] + b[i];
    if (OUT_BF16) outb[(size_t)row * dim + i] = f2bf(v);
    else          outf[(size_t)row * dim + i] = v;
  }
}

// ---------------------------------------------------------------------------
// bf16 GEMM: out[M,N] = epilogue( A[M,K] @ W[K,N] ).
// 256 threads = 8 waves (2 x 4). Block tile 128x128, K step 32.
// Wave tile 64x32 = 4x2 WMMA accumulators. Double-buffered LDS; A tile via
// async global->LDS b128 copies, W tile transposed into [n][k] manually.
// ---------------------------------------------------------------------------
#define BM   128
#define BN   128
#define BK   32
#define LSTR 40   // LDS row stride in shorts: 80B = 16B aligned, bank-spread

template <int HAS_BIAS, int HAS_RES, int GELU_ACT, int OUT_BF16>
__global__ __launch_bounds__(256) void gemm_bf16_kernel(
    const unsigned short* __restrict__ A, const unsigned short* __restrict__ W,
    const float* __restrict__ bias, const float* __restrict__ res,
    float* __restrict__ outf, unsigned short* __restrict__ outb,
    int M, int N, int K, int ldw, int ldo) {
  __shared__ unsigned short As[2][BM * LSTR];
  __shared__ unsigned short Bs[2][BN * LSTR];
  const int tid   = threadIdx.x;
  const int lane  = tid & 31, wid = tid >> 5;
  const int wm    = wid & 1,  wn  = wid >> 1;   // 2 x 4 wave grid
  const int m0    = blockIdx.y * BM, n0 = blockIdx.x * BN;
  const int row16 = lane & 15, half = lane >> 4;

  v8f acc[4][2];
  #pragma unroll
  for (int i = 0; i < 4; ++i)
    #pragma unroll
    for (int j = 0; j < 2; ++j) acc[i][j] = vzero8();

  auto stage_tiles = [&](int k0, int bufi) {
    // A tile 128x32 bf16: straight async copy, 16B per lane per issue.
    #pragma unroll
    for (int it = 0; it < 2; ++it) {
      int i = tid + it * 256;
      int r = i >> 2, seg = i & 3;
      async_copy_b128(&As[bufi][r * LSTR + seg * 8],
                      A + (size_t)(m0 + r) * K + k0 + seg * 8);
    }
    // W tile 32x128 bf16, transposed into [n][k] so B-fragment K-pairs are
    // contiguous (layout per ISA 7.12.2).
    #pragma unroll
    for (int it = 0; it < 2; ++it) {
      int i = tid + it * 256;
      int kr = i >> 4, seg = i & 15;
      uint4 val = *reinterpret_cast<const uint4*>(
          W + (size_t)(k0 + kr) * ldw + n0 + seg * 8);
      const unsigned short* ps = reinterpret_cast<const unsigned short*>(&val);
      #pragma unroll
      for (int j = 0; j < 8; ++j) Bs[bufi][(seg * 8 + j) * LSTR + kr] = ps[j];
    }
  };

  stage_tiles(0, 0);
  int buf = 0;
  for (int k0 = 0; k0 < K; k0 += BK) {
    wait_async0();          // our async A copies for `buf` are in LDS
    __syncthreads();        // everyone's A+B stores for `buf` visible
    if (k0 + BK < K) stage_tiles(k0 + BK, buf ^ 1);  // prefetch next tile

    Frag af[4], bfr[2];
    #pragma unroll
    for (int i = 0; i < 4; ++i) {
      const uint4* p4 = reinterpret_cast<const uint4*>(
          &As[buf][(wm * 64 + i * 16 + row16) * LSTR]);
      af[i].q[0] = p4[half];       // K = half*8 .. +7
      af[i].q[1] = p4[half + 2];   // K = half*8+16 .. +23
    }
    #pragma unroll
    for (int j = 0; j < 2; ++j) {
      const uint4* p4 = reinterpret_cast<const uint4*>(
          &Bs[buf][(wn * 32 + j * 16 + row16) * LSTR]);
      bfr[j].q[0] = p4[half];      // K = half*8 .. +7
      bfr[j].q[1] = p4[half + 2];  // K = half*8+16 .. +23
    }
    #pragma unroll
    for (int i = 0; i < 4; ++i)
      #pragma unroll
      for (int j = 0; j < 2; ++j)
        acc[i][j] = wmma_bf(af[i], bfr[j], acc[i][j]);
    buf ^= 1;
  }

  // Epilogue. C layout: VGPR r, lane l -> (M = r + 8*(l/16), N = l%16).
  #pragma unroll
  for (int i = 0; i < 4; ++i) {
    #pragma unroll
    for (int j = 0; j < 2; ++j) {
      #pragma unroll
      for (int r = 0; r < 8; ++r) {
        int m = m0 + wm * 64 + i * 16 + r + half * 8;
        int n = n0 + wn * 32 + j * 16 + row16;
        float v = acc[i][j][r];
        if (HAS_BIAS) v += bias[n];
        if (HAS_RES)  v += res[(size_t)m * ldo + n];
        if (GELU_ACT) v = 0.5f * v * (1.0f + erff(v * 0.70710678118654752f));
        if (OUT_BF16) outb[(size_t)m * ldo + n] = f2bf(v);
        else          outf[(size_t)m * ldo + n] = v;
      }
    }
  }
}

// ---------------------------------------------------------------------------
// Flash attention with additive bias-dots.
// dots = (Q.K + Qb.Kb) * scale; softmax streamed over 32-key steps.
// 128 threads = 4 waves; wave owns a 16-row query tile; block covers 64 rows.
// qkv:  [MROWS, 3072] bf16 (q | k | v), bqkv: [MROWS, 2048] bf16 (qb | kb).
// ---------------------------------------------------------------------------
__global__ __launch_bounds__(128) void attn_kernel(
    const unsigned short* __restrict__ qkv,
    const unsigned short* __restrict__ bqkv,
    unsigned short* __restrict__ attno) {
  __shared__ unsigned short Pl[4 * 16 * LSTR];
  const int bh   = blockIdx.y;
  const int b    = bh >> 4, h = bh & 15;
  const int tid  = threadIdx.x;
  const int lane = tid & 31, wave = tid >> 5;
  const int row16 = lane & 15, half = lane >> 4;
  const int mrow = b * TLEN + blockIdx.x * 64 + wave * 16;  // tile base row
  const float scale = 0.125f;                               // 64^-0.5

  // Q / Qbias A-fragments over d = 0..63 (two K=32 fragments each).
  Frag qlo, qhi, qblo, qbhi;
  {
    const uint4* qp4 = reinterpret_cast<const uint4*>(
        qkv + (size_t)(mrow + row16) * 3072 + h * 64);
    const uint4* qb4 = reinterpret_cast<const uint4*>(
        bqkv + (size_t)(mrow + row16) * 2048 + h * 64);
    qlo.q[0]  = qp4[half];     qlo.q[1]  = qp4[half + 2];
    qhi.q[0]  = qp4[half + 4]; qhi.q[1]  = qp4[half + 6];
    qblo.q[0] = qb4[half];     qblo.q[1] = qb4[half + 2];
    qbhi.q[0] = qb4[half + 4]; qbhi.q[1] = qb4[half + 6];
  }

  float run_m[8], run_s[8];
  v8f o[4];
  #pragma unroll
  for (int r = 0; r < 8; ++r) { run_m[r] = -3.0e38f; run_s[r] = 0.f; }
  #pragma unroll
  for (int d = 0; d < 4; ++d) o[d] = vzero8();

  unsigned short* pw = &Pl[wave * 16 * LSTR];

  for (int jt = 0; jt < TLEN; jt += 32) {
    // S tiles: 16 queries x 32 keys.
    v8f s[2];
    #pragma unroll
    for (int t = 0; t < 2; ++t) {
      s[t] = vzero8();
      const int krow = b * TLEN + jt + t * 16 + row16;
      const uint4* kp4 = reinterpret_cast<const uint4*>(
          qkv + (size_t)krow * 3072 + 1024 + h * 64);
      const uint4* kb4 = reinterpret_cast<const uint4*>(
          bqkv + (size_t)krow * 2048 + 1024 + h * 64);
      Frag kflo, kfhi, kbflo, kbfhi;
      kflo.q[0]  = kp4[half];     kflo.q[1]  = kp4[half + 2];
      kfhi.q[0]  = kp4[half + 4]; kfhi.q[1]  = kp4[half + 6];
      kbflo.q[0] = kb4[half];     kbflo.q[1] = kb4[half + 2];
      kbfhi.q[0] = kb4[half + 4]; kbfhi.q[1] = kb4[half + 6];
      s[t] = wmma_bf(qlo,  kflo,  s[t]);
      s[t] = wmma_bf(qhi,  kfhi,  s[t]);
      s[t] = wmma_bf(qblo, kbflo, s[t]);
      s[t] = wmma_bf(qbhi, kbfhi, s[t]);
    }

    // Online softmax; row stats reduced across the 16-lane C-layout groups.
    #pragma unroll
    for (int r = 0; r < 8; ++r) {
      float a0 = s[0][r] * scale, a1 = s[1][r] * scale;
      float mx = fmaxf(a0, a1);
      #pragma unroll
      for (int off = 8; off; off >>= 1) mx = fmaxf(mx, __shfl_xor(mx, off, 32));
      float nm  = fmaxf(run_m[r], mx);
      float fac = __expf(run_m[r] - nm);
      run_m[r] = nm;
      float p0 = __expf(a0 - nm), p1 = __expf(a1 - nm);
      float ls = p0 + p1;
      #pragma unroll
      for (int off = 8; off; off >>= 1) ls += __shfl_xor(ls, off, 32);
      run_s[r] = run_s[r] * fac + ls;
      #pragma unroll
      for (int d = 0; d < 4; ++d) o[d][r] *= fac;
      const int prow = r + half * 8;
      pw[prow * LSTR + row16]      = f2bf(p0);
      pw[prow * LSTR + 16 + row16] = f2bf(p1);
    }
    // Within-wave LDS RAW: wait on DScnt before re-reading P as A-fragment.
    asm volatile("s_wait_dscnt 0" ::: "memory");

    Frag pf;  // P as 16x32 bf16 A-fragment (K = 32 keys)
    {
      const uint4* pp4 = reinterpret_cast<const uint4*>(pw + row16 * LSTR);
      pf.q[0] = pp4[half * 2];
      pf.q[1] = pp4[half * 2 + 1];
    }
    // O += P @ V over the 32-key step, 4 d-tiles of 16.
    #pragma unroll
    for (int dt = 0; dt < 4; ++dt) {
      Frag vf;
      const int kb2  = half * 16;
      const int dcol = 2048 + h * 64 + dt * 16 + row16;
      #pragma unroll
      for (int j = 0; j < 8; ++j) {
        const int k2 = jt + kb2 + 2 * j;
        unsigned lo = qkv[(size_t)(b * TLEN + k2) * 3072 + dcol];
        unsigned hi = qkv[(size_t)(b * TLEN + k2 + 1) * 3072 + dcol];
        vf.u[j] = lo | (hi << 16);
      }
      o[dt] = wmma_bf(pf, vf, o[dt]);
    }
  }

  // Normalize and store bf16 attention output [MROWS, 1024].
  #pragma unroll
  for (int dt = 0; dt < 4; ++dt) {
    #pragma unroll
    for (int r = 0; r < 8; ++r) {
      int m = mrow + r + half * 8;
      attno[(size_t)m * 1024 + h * 64 + dt * 16 + row16] =
          f2bf(o[dt][r] / run_s[r]);
    }
  }
}

// ---------------------------------------------------------------------------
// Host orchestration
// ---------------------------------------------------------------------------
extern "C" void kernel_launch(void* const* d_in, const int* in_sizes, int n_in,
                              void* d_out, int out_size, void* d_ws,
                              size_t ws_size, hipStream_t stream) {
  (void)in_sizes; (void)n_in; (void)out_size; (void)ws_size;
  const float* x_in = (const float*)d_in[0];
  const float* bias = (const float*)d_in[1];
  const float* wqkv = (const float*)d_in[2];
  const float* wout = (const float*)d_in[3];
  const float* bout = (const float*)d_in[4];
  const float* ln1g = (const float*)d_in[5];
  const float* ln1b = (const float*)d_in[6];
  const float* w1   = (const float*)d_in[7];
  const float* b1   = (const float*)d_in[8];
  const float* w2   = (const float*)d_in[9];
  const float* b2   = (const float*)d_in[10];
  const float* ln2g = (const float*)d_in[11];
  const float* ln2b = (const float*)d_in[12];
  const float* lnfg = (const float*)d_in[13];
  const float* lnfb = (const float*)d_in[14];

  float* x = (float*)d_out;  // residual stream lives in d_out; final LN in-place

  char* wsb = (char*)d_ws;
  size_t off = 0;
  auto alloc = [&](size_t elems) {
    unsigned short* p = (unsigned short*)(wsb + off);
    off += (elems * 2 + 255) & ~(size_t)255;
    return p;
  };
  unsigned short* hbf    = alloc((size_t)MROWS * DIMC);   // LN output (bf16)
  unsigned short* biasbf = alloc((size_t)MROWS * DIMC);   // bias input (bf16)
  unsigned short* qkv    = alloc((size_t)MROWS * 3072);   // q|k|v
  unsigned short* bqkv   = alloc((size_t)MROWS * 2048);   // qb|kb
  unsigned short* attno  = alloc((size_t)MROWS * DIMC);   // attention out
  unsigned short* ffb    = alloc((size_t)MROWS * FFDIM);  // gelu(h@w1+b1)
  unsigned short* wqkvb  = alloc((size_t)DIMC * 3072);    // per-layer weights
  unsigned short* woutb  = alloc((size_t)DIMC * DIMC);
  unsigned short* w1b    = alloc((size_t)DIMC * FFDIM);
  unsigned short* w2b    = alloc((size_t)FFDIM * DIMC);

  hipMemcpyAsync(x, x_in, (size_t)MROWS * DIMC * sizeof(float),
                 hipMemcpyDeviceToDevice, stream);

  auto cvt = [&](const float* src, unsigned short* dst, size_t n) {
    int blocks = (int)((n + 255) / 256);
    convert_f32_bf16<<<blocks, 256, 0, stream>>>(src, dst, (int)n);
  };
  cvt(bias, biasbf, (size_t)MROWS * DIMC);

  for (int l = 0; l < DEPTHC; ++l) {
    cvt(wqkv + (size_t)l * DIMC * 3072, wqkvb, (size_t)DIMC * 3072);
    layernorm_kernel<1><<<MROWS, 256, 0, stream>>>(
        x, ln1g + l * DIMC, ln1b + l * DIMC, nullptr, hbf, DIMC);
    // qkv = ln1(x) @ wqkv  (bf16 out)
    gemm_bf16_kernel<0, 0, 0, 1><<<dim3(3072 / BN, MROWS / BM), 256, 0, stream>>>(
        hbf, wqkvb, nullptr, nullptr, nullptr, qkv, MROWS, 3072, DIMC, 3072, 3072);
    // bias-qkv: only q,k halves needed (N = 2048)
    gemm_bf16_kernel<0, 0, 0, 1><<<dim3(2048 / BN, MROWS / BM), 256, 0, stream>>>(
        biasbf, wqkvb, nullptr, nullptr, nullptr, bqkv, MROWS, 2048, DIMC, 3072, 2048);
    attn_kernel<<<dim3(TLEN / 64, BATCH * NHEAD), 128, 0, stream>>>(qkv, bqkv, attno);
    // x = x + attno @ wout + bout   (fp32 out)
    cvt(wout + (size_t)l * DIMC * DIMC, woutb, (size_t)DIMC * DIMC);
    gemm_bf16_kernel<1, 1, 0, 0><<<dim3(DIMC / BN, MROWS / BM), 256, 0, stream>>>(
        attno, woutb, bout + l * DIMC, x, x, nullptr, MROWS, DIMC, DIMC, DIMC, DIMC);
    // FF
    layernorm_kernel<1><<<MROWS, 256, 0, stream>>>(
        x, ln2g + l * DIMC, ln2b + l * DIMC, nullptr, hbf, DIMC);
    cvt(w1 + (size_t)l * DIMC * FFDIM, w1b, (size_t)DIMC * FFDIM);
    gemm_bf16_kernel<1, 0, 1, 1><<<dim3(FFDIM / BN, MROWS / BM), 256, 0, stream>>>(
        hbf, w1b, b1 + l * FFDIM, nullptr, nullptr, ffb, MROWS, FFDIM, DIMC, FFDIM, FFDIM);
    cvt(w2 + (size_t)l * FFDIM * DIMC, w2b, (size_t)FFDIM * DIMC);
    gemm_bf16_kernel<1, 1, 0, 0><<<dim3(DIMC / BN, MROWS / BM), 256, 0, stream>>>(
        ffb, w2b, b2 + l * DIMC, x, x, nullptr, MROWS, DIMC, FFDIM, DIMC, DIMC);
  }
  // Final LayerNorm, fp32 in-place on d_out.
  layernorm_kernel<0><<<MROWS, 256, 0, stream>>>(x, lnfg, lnfb, x, nullptr, DIMC);
}
